// SelfAttentionModule_65481071404660
// MI455X (gfx1250) — matrix operand order repaired and checked by reading, not compiled, not run on hardware
//
#include <hip/hip_runtime.h>
#include <hip/hip_bf16.h>

typedef __bf16 bf16_t;
typedef __attribute__((ext_vector_type(16))) __bf16 v16bf;
typedef __attribute__((ext_vector_type(8)))  float  v8f;
typedef unsigned int u32;
typedef u32   u32x4 __attribute__((ext_vector_type(4)));
typedef float f32x4 __attribute__((ext_vector_type(4)));

union FragU { u32x4 u[2]; v16bf v; };

// ---------------------------------------------------------------------------
// Fragment loaders (wave32 WMMA 16x16x32 bf16 layouts, CDNA5 ISA 7.12.2)
// A: 16xK row-major.  lane<16: row=lane, K=kb..kb+7 then kb+16..kb+23 (kb=0)
//                     lane>=16: same row, kb=8
// B: loaded from Bt ([N,K] row-major).  lane<16: col=lane, K=0..7 then 16..23
//                                       lane>=16: col=lane-16, K=8..15 then 24..31
// ---------------------------------------------------------------------------
__device__ inline v16bf load_a_frag(const bf16_t* __restrict__ Arow, int lda, int k0) {
    const int lane = threadIdx.x & 31;
    const int m    = lane & 15;
    const int kb   = (lane & 16) ? 8 : 0;
    const bf16_t* p = Arow + (long long)m * lda + k0 + kb;
    FragU f;
    f.u[0] = *(const u32x4*)(p);        // K = kb .. kb+7
    f.u[1] = *(const u32x4*)(p + 16);   // K = kb+16 .. kb+23
    return f.v;
}

__device__ inline v16bf load_bt_frag(const bf16_t* __restrict__ Btrow, int ldb, int k0) {
    const int lane = threadIdx.x & 31;
    const int n    = lane & 15;
    const int kb   = (lane & 16) ? 8 : 0;
    const bf16_t* p = Btrow + (long long)n * ldb + k0 + kb;
    FragU f;
    f.u[0] = *(const u32x4*)(p);        // K = kb .. kb+7
    f.u[1] = *(const u32x4*)(p + 16);   // K = kb+16 .. kb+23
    return f.v;
}

// ---------------------------------------------------------------------------
// Generic bf16 WMMA GEMM:  C[m,n] = (sum_k A[m,k]*Bt[n,k] + bias[n]) * scale
// Block: 256 threads = 8 waves (2 along M x 4 along N).
// Wave tile: 32x64 = 2 A-frags x 4 B-frags -> 8 WMMAs per 12 b128 loads.
// Block tile: 64x256.  grid = (N/256, Mbatch/64, nbatch).
// TRANS_OUT stores C as [N, Mbatch] (contiguous 8-element stores per lane).
// ---------------------------------------------------------------------------
template <typename OUT_T, bool TRANS_OUT>
__global__ __launch_bounds__(256)
void gemm_bf16_wmma(const bf16_t* __restrict__ A,  long long aBatch, int lda,
                    const bf16_t* __restrict__ Bt, long long bBatch, int ldb,
                    OUT_T* __restrict__ C,         long long cBatch, int ldc,
                    const float* __restrict__ bias, float scale, int K) {
    const int z = blockIdx.z;
    const bf16_t* Abase = A  + (long long)z * aBatch;
    const bf16_t* Bbase = Bt + (long long)z * bBatch;
    OUT_T*        Cb    = C  + (long long)z * cBatch;

    const int wave = threadIdx.x >> 5;
    const int wm   = wave >> 2;                  // 0..1
    const int wn   = wave & 3;                   // 0..3
    const int m0   = blockIdx.y * 64  + wm * 32;
    const int n0   = blockIdx.x * 256 + wn * 64;

    const bf16_t* Arow0 = Abase + (long long)(m0 +  0) * lda;
    const bf16_t* Arow1 = Abase + (long long)(m0 + 16) * lda;
    const bf16_t* Brow0 = Bbase + (long long)(n0 +  0) * ldb;
    const bf16_t* Brow1 = Bbase + (long long)(n0 + 16) * ldb;
    const bf16_t* Brow2 = Bbase + (long long)(n0 + 32) * ldb;
    const bf16_t* Brow3 = Bbase + (long long)(n0 + 48) * ldb;

    v8f acc[2][4];
#pragma unroll
    for (int i = 0; i < 2; ++i)
#pragma unroll
        for (int j = 0; j < 4; ++j)
            acc[i][j] = (v8f){0,0,0,0,0,0,0,0};

    for (int k0 = 0; k0 < K; k0 += 32) {
        v16bf a0 = load_a_frag(Arow0, lda, k0);
        v16bf a1 = load_a_frag(Arow1, lda, k0);
        v16bf b0 = load_bt_frag(Brow0, ldb, k0);
        v16bf b1 = load_bt_frag(Brow1, ldb, k0);
        v16bf b2 = load_bt_frag(Brow2, ldb, k0);
        v16bf b3 = load_bt_frag(Brow3, ldb, k0);
        acc[0][0] = __builtin_amdgcn_wmma_f32_16x16x32_bf16(false, a0, false, b0, (short)0, acc[0][0], false, false);
        acc[0][1] = __builtin_amdgcn_wmma_f32_16x16x32_bf16(false, a0, false, b1, (short)0, acc[0][1], false, false);
        acc[0][2] = __builtin_amdgcn_wmma_f32_16x16x32_bf16(false, a0, false, b2, (short)0, acc[0][2], false, false);
        acc[0][3] = __builtin_amdgcn_wmma_f32_16x16x32_bf16(false, a0, false, b3, (short)0, acc[0][3], false, false);
        acc[1][0] = __builtin_amdgcn_wmma_f32_16x16x32_bf16(false, a1, false, b0, (short)0, acc[1][0], false, false);
        acc[1][1] = __builtin_amdgcn_wmma_f32_16x16x32_bf16(false, a1, false, b1, (short)0, acc[1][1], false, false);
        acc[1][2] = __builtin_amdgcn_wmma_f32_16x16x32_bf16(false, a1, false, b2, (short)0, acc[1][2], false, false);
        acc[1][3] = __builtin_amdgcn_wmma_f32_16x16x32_bf16(false, a1, false, b3, (short)0, acc[1][3], false, false);
    }

    // C/D layout: lane<16 -> (M=r, N=lane); lane>=16 -> (M=8+r, N=lane-16)
    const int lane = threadIdx.x & 31;
    const int n    = lane & 15;
    const int mb   = (lane & 16) ? 8 : 0;

#pragma unroll
    for (int i = 0; i < 2; ++i) {
        const int mrow = m0 + i * 16 + mb;
#pragma unroll
        for (int j = 0; j < 4; ++j) {
            const int col = n0 + j * 16 + n;
            const float badd = bias ? bias[col] : 0.0f;
            if (!TRANS_OUT) {
#pragma unroll
                for (int r = 0; r < 8; ++r) {
                    float v = (acc[i][j][r] + badd) * scale;
                    Cb[(long long)(mrow + r) * ldc + col] = (OUT_T)v;
                }
            } else {
                // transposed store: C[col][mrow .. mrow+7] contiguous per lane
                OUT_T tmp[8];
#pragma unroll
                for (int r = 0; r < 8; ++r)
                    tmp[r] = (OUT_T)((acc[i][j][r] + badd) * scale);
                OUT_T* p = Cb + (long long)col * ldc + mrow;
#pragma unroll
                for (int r = 0; r < 8; ++r) p[r] = tmp[r];
            }
        }
    }
}

// ---------------------------------------------------------------------------
// Row softmax over 2048 f32, writing bf16 probs in place (row stride 4096 bf16)
// ---------------------------------------------------------------------------
__global__ __launch_bounds__(256)
void softmax_rows_2048(float* __restrict__ scores) {
    const long long row = blockIdx.x;
    float* srow = scores + row * 2048;
    bf16_t* prow = (bf16_t*)srow;              // in-place bf16 (first 4096 bytes)

    const int t = threadIdx.x;
    f32x4 v0 = *(const f32x4*)(srow + t * 8);
    f32x4 v1 = *(const f32x4*)(srow + t * 8 + 4);
    float v[8] = {v0.x, v0.y, v0.z, v0.w, v1.x, v1.y, v1.z, v1.w};

    __shared__ float red[256];
    float m = v[0];
#pragma unroll
    for (int i = 1; i < 8; ++i) m = fmaxf(m, v[i]);
    red[t] = m;
    __syncthreads();
    for (int s = 128; s > 0; s >>= 1) {
        if (t < s) red[t] = fmaxf(red[t], red[t + s]);
        __syncthreads();
    }
    const float rmax = red[0];
    __syncthreads();

    float sum = 0.0f;
#pragma unroll
    for (int i = 0; i < 8; ++i) { v[i] = __expf(v[i] - rmax); sum += v[i]; }
    red[t] = sum;
    __syncthreads();
    for (int s = 128; s > 0; s >>= 1) {
        if (t < s) red[t] += red[t + s];
        __syncthreads();
    }
    const float inv = 1.0f / red[0];
    __syncthreads();                           // all reads done before any write

    union { bf16_t h[8]; u32x4 u; } o;
#pragma unroll
    for (int i = 0; i < 8; ++i) o.h[i] = (bf16_t)(v[i] * inv);
    *(u32x4*)(prow + t * 8) = o.u;
}

// ---------------------------------------------------------------------------
// Converters
// ---------------------------------------------------------------------------
__global__ __launch_bounds__(256)
void cvt_f32_to_bf16(const float* __restrict__ in, bf16_t* __restrict__ out, int n) {
    int i = blockIdx.x * 256 + threadIdx.x;
    if (i < n) out[i] = (bf16_t)in[i];
}

// out[n][k] = in[k][n] for 1024x1024 (weights -> transposed bf16)
__global__ __launch_bounds__(256)
void transpose_cvt_1024(const float* __restrict__ in, bf16_t* __restrict__ out) {
    int idx = blockIdx.x * 256 + threadIdx.x;   // flat over outputs, coalesced writes
    int n = idx >> 10;
    int k = idx & 1023;
    out[idx] = (bf16_t)in[k * 1024 + n];
}

// ---------------------------------------------------------------------------
// Launch: x,wq,bq,wk,bk,wv,bv,wo,bo  ->  out [4,2048,1024] f32
// ---------------------------------------------------------------------------
extern "C" void kernel_launch(void* const* d_in, const int* in_sizes, int n_in,
                              void* d_out, int out_size, void* d_ws, size_t ws_size,
                              hipStream_t stream) {
    (void)in_sizes; (void)n_in; (void)out_size; (void)ws_size;

    const float* x  = (const float*)d_in[0];
    const float* wq = (const float*)d_in[1];
    const float* bq = (const float*)d_in[2];
    const float* wk = (const float*)d_in[3];
    const float* bk = (const float*)d_in[4];
    const float* wv = (const float*)d_in[5];
    const float* bv = (const float*)d_in[6];
    const float* wo = (const float*)d_in[7];
    const float* bo = (const float*)d_in[8];
    float* out = (float*)d_out;

    const int D = 1024, S = 2048, B = 4;
    const int M = B * S;                         // 8192

    char* ws = (char*)d_ws;
    bf16_t* xb     = (bf16_t*)(ws + 0);                  // 16 MiB [8192,1024]
    bf16_t* wqt    = (bf16_t*)(ws + 16777216);           //  2 MiB [N,K]
    bf16_t* wkt    = (bf16_t*)(ws + 18874368);
    bf16_t* wvt    = (bf16_t*)(ws + 20971520);
    bf16_t* wot    = (bf16_t*)(ws + 23068672);
    bf16_t* Qb     = (bf16_t*)(ws + 25165824);           // 16 MiB [8192,1024]
    bf16_t* Kb     = (bf16_t*)(ws + 41943040);           // 16 MiB [8192,1024]
    bf16_t* Vt     = (bf16_t*)(ws + 58720256);           // 16 MiB [4][1024,2048]
    float*  scores = (float*) (ws + 75497472);           // 64 MiB [4][2048,2048]
    bf16_t* Pb     = (bf16_t*)scores;                    // in-place probs, ld=4096
    bf16_t* Ab     = xb;                                 // reuse xb after QKV done

    // 1) conversions
    cvt_f32_to_bf16<<<(M * D) / 256, 256, 0, stream>>>(x, xb, M * D);
    transpose_cvt_1024<<<(D * D) / 256, 256, 0, stream>>>(wq, wqt);
    transpose_cvt_1024<<<(D * D) / 256, 256, 0, stream>>>(wk, wkt);
    transpose_cvt_1024<<<(D * D) / 256, 256, 0, stream>>>(wv, wvt);
    transpose_cvt_1024<<<(D * D) / 256, 256, 0, stream>>>(wo, wot);

    // 2) Q = (x@wq + bq) * 1/sqrt(D)   (scale folded into Q)
    gemm_bf16_wmma<bf16_t, false><<<dim3(D / 256, M / 64, 1), 256, 0, stream>>>(
        xb, 0, D, wqt, 0, D, Qb, 0, D, bq, 0.03125f, D);
    // 3) K = x@wk + bk
    gemm_bf16_wmma<bf16_t, false><<<dim3(D / 256, M / 64, 1), 256, 0, stream>>>(
        xb, 0, D, wkt, 0, D, Kb, 0, D, bk, 1.0f, D);
    // 4) Vt[b] = (x@wv + bv)^T per batch  ([D, S])
    gemm_bf16_wmma<bf16_t, true><<<dim3(D / 256, S / 64, B), 256, 0, stream>>>(
        xb, (long long)S * D, D, wvt, 0, D, Vt, (long long)D * S, S, bv, 1.0f, D);

    // 5) scores[b] = Q[b] @ K[b]^T   (K row-major acts as Bt)
    gemm_bf16_wmma<float, false><<<dim3(S / 256, S / 64, B), 256, 0, stream>>>(
        Qb, (long long)S * D, D, Kb, (long long)S * D, D,
        scores, (long long)S * S, S, nullptr, 1.0f, D);

    // 6) softmax rows -> bf16 probs in place (row stride 4096 bf16)
    softmax_rows_2048<<<M, 256, 0, stream>>>(scores);

    // 7) attn_out[b] = P[b] @ V[b]   (Vt is Bt)
    gemm_bf16_wmma<bf16_t, false><<<dim3(D / 256, S / 64, B), 256, 0, stream>>>(
        Pb, (long long)S * 4096, 4096, Vt, (long long)D * S, S,
        Ab, (long long)S * D, D, nullptr, 1.0f, S);

    // 8) out = attn_out @ wo + bo   (f32 output)
    gemm_bf16_wmma<float, false><<<dim3(D / 256, M / 64, 1), 256, 0, stream>>>(
        Ab, 0, D, wot, 0, D, out, 0, D, bo, 1.0f, D);
}
